// MoeBlock_RS_37228776522470
// MI455X (gfx1250) — matrix operand order repaired and checked by reading, not compiled, not run on hardware
//
#include <hip/hip_runtime.h>
#include <hip/hip_bf16.h>

// ---------------- problem constants ----------------
constexpr int Bn = 4;
constexpr int Sn = 4096;
constexpr int H  = 2048;
constexpr int F  = 256;
constexpr int E  = 12;
constexpr int T  = Bn * Sn;            // 16384 tokens
constexpr int CAP = 1707;              // ceil(T/E * 1.25)
constexpr int TILES = (CAP + 15) / 16; // 107 token-tiles per expert

// ---------------- vector types ----------------
typedef __attribute__((ext_vector_type(16))) __bf16          v16bf;
typedef __attribute__((ext_vector_type(8)))  float           v8f;
typedef __attribute__((ext_vector_type(16))) unsigned short  v16us;
typedef __attribute__((ext_vector_type(8)))  unsigned short  v8us;
typedef __attribute__((ext_vector_type(4)))  unsigned int    v4u;
typedef __attribute__((ext_vector_type(8)))  int             v8i_t;
typedef __attribute__((ext_vector_type(4)))  int             v4i_t;

// fp32 -> bf16 bits, round-to-nearest-even
__device__ __forceinline__ unsigned short f2bf(float f) {
    unsigned u = __float_as_uint(f);
    u += 0x7fffu + ((u >> 16) & 1u);
    return (unsigned short)(u >> 16);
}

// A-fragment (16x32 bf16) from an LDS row (contiguous k order, same as B):
// lane M = lane&15; chunk c = 8*(lane>=16); Ks = [kb+2c,kb+2c+16)
__device__ __forceinline__ v16bf load_a_frag(const unsigned short* row, int kb, int chunk) {
    v8us lo = *(const v8us*)(row + kb + 2 * chunk);
    v8us hi = *(const v8us*)(row + kb + 2 * chunk + 8);
    v16us c = __builtin_shufflevector(lo, hi, 0,1,2,3,4,5,6,7,8,9,10,11,12,13,14,15);
    return __builtin_bit_cast(v16bf, c);
}

// B-fragment (32x16): per lane N = lane&15, K = 16*(lane>=16) + [0,16)
// weights are pre-converted bf16 -> 16 contiguous bf16 = 32B, two b128 loads, no cvt VALU
__device__ __forceinline__ v16bf load_b_frag(const unsigned short* p) {
    return __builtin_bit_cast(v16bf, *(const v16us*)p);
}

__device__ __forceinline__ v8f wmma_bf16(v16bf a, v16bf b, v8f c) {
    return __builtin_amdgcn_wmma_f32_16x16x32_bf16(false, a, false, b, (short)0, c, false, false);
}

// ---------------- kernel 0: fp32 -> bf16 bulk convert ----------------
__global__ __launch_bounds__(256) void cvt_bf16_kernel(
    const float* __restrict__ in, unsigned short* __restrict__ out) {
    const long long i = ((long long)blockIdx.x * 256 + threadIdx.x) * 8;
    float4 a = *(const float4*)(in + i);
    float4 b = *(const float4*)(in + i + 4);
    v8us r;
    r[0] = f2bf(a.x); r[1] = f2bf(a.y); r[2] = f2bf(a.z); r[3] = f2bf(a.w);
    r[4] = f2bf(b.x); r[5] = f2bf(b.y); r[6] = f2bf(b.z); r[7] = f2bf(b.w);
    *(v8us*)(out + i) = r;
}

// ---------------- kernel 1: gating (top-2, renormalized) ----------------
__global__ __launch_bounds__(256) void gate_kernel(
    const float* __restrict__ x, const float* __restrict__ gw,
    int* __restrict__ sel, float* __restrict__ rw) {
    const int lane = threadIdx.x & 31, wid = threadIdx.x >> 5;
    const int tok = blockIdx.x * 8 + wid;
    const float* xr = x + (size_t)tok * H;
    float acc[E];
#pragma unroll
    for (int e = 0; e < E; ++e) acc[e] = 0.f;
    for (int h = lane; h < H; h += 32) {
        float xv = xr[h];
#pragma unroll
        for (int e = 0; e < E; ++e) acc[e] += xv * gw[e * H + h];
    }
#pragma unroll
    for (int e = 0; e < E; ++e) {
        float v = acc[e];
        for (int off = 16; off > 0; off >>= 1) v += __shfl_xor(v, off);
        acc[e] = v;
    }
    if (lane == 0) {
        int i1 = 0; float l1 = acc[0];
#pragma unroll
        for (int e = 1; e < E; ++e) if (acc[e] > l1) { l1 = acc[e]; i1 = e; }
        int i2 = 0; float l2 = -3.4e38f;
#pragma unroll
        for (int e = 0; e < E; ++e) if (e != i1 && acc[e] > l2) { l2 = acc[e]; i2 = e; }
        float p1 = 1.f / (1.f + __expf(l2 - l1));   // renormalized top-2 weight
        sel[tok * 2 + 0] = i1; sel[tok * 2 + 1] = i2;
        rw [tok * 2 + 0] = p1; rw [tok * 2 + 1] = 1.f - p1;
    }
}

// ---------------- kernel 2: deterministic capacity routing ----------------
__global__ __launch_bounds__(1024) void route_kernel(
    const int* __restrict__ sel, const float* __restrict__ rw,
    int* __restrict__ etok, float* __restrict__ ew, int* __restrict__ ecnt) {
    const int e = blockIdx.x;
    const int tid = threadIdx.x, lane = tid & 31, wid = tid >> 5;
    __shared__ int wsum[32];
    __shared__ int wbase[32];
    __shared__ int s_total;
    int running = 0;
    for (int chunk = 0; chunk < 2 * T; chunk += 1024) {
        int f = chunk + tid;
        int k = (f >= T) ? 1 : 0;
        int t = f - k * T;
        bool m = (sel[t * 2 + k] == e);
        unsigned bal = (unsigned)__ballot(m);
        int lanePre = __popc(bal & ((1u << lane) - 1u));
        if (lane == 0) wsum[wid] = __popc(bal);
        __syncthreads();
        if (wid == 0) {
            int v = wsum[lane];
            int incl = v;
            for (int off = 1; off < 32; off <<= 1) {
                int n = __shfl_up(incl, off);
                if (lane >= off) incl += n;
            }
            wbase[lane] = incl - v;
            if (lane == 31) s_total = incl;
        }
        __syncthreads();
        int pos = running + wbase[wid] + lanePre;
        if (m && pos < CAP) {
            etok[e * CAP + pos] = t;
            ew  [e * CAP + pos] = rw[t * 2 + k];
        }
        running += s_total;
        __syncthreads();
    }
    if (tid == 0) ecnt[e] = running < CAP ? running : CAP;
}

// ---------------- kernel 3: expert FFN, BF16 WMMA + TDM gather ----------------
// 256 threads = 8 waves; one 16-token tile per block; x panels fetched by the
// Tensor Data Mover in gather mode (16 row indices = our 16 token ids),
// double-buffered in LDS and synchronized with s_wait_tensorcnt.
__global__ __launch_bounds__(256) void ffn_kernel(
    const unsigned short* __restrict__ xbf,
    const unsigned short* __restrict__ w1bf,
    const unsigned short* __restrict__ w2bf,
    const unsigned short* __restrict__ w3bf,
    const int* __restrict__ etok, const float* __restrict__ ew,
    const int* __restrict__ ecnt, float* __restrict__ out) {
    const int e    = blockIdx.x / TILES;
    const int tile = blockIdx.x % TILES;
    const int cnt  = ecnt[e];
    if (tile * 16 >= cnt) return;

    __shared__ unsigned short lx[2][16 * 512]; // double-buffered x K-panels (2x16KB)
    __shared__ unsigned short lh[16 * 256];    // intermediate h (8KB)
    __shared__ int   s_tok[16];
    __shared__ float s_wgt[16];

    const int tid = threadIdx.x, lane = tid & 31, wid = tid >> 5;
    const int hw = lane >> 4;           // half-wave
    const int l15 = lane & 15;
    const int chunk = hw * 8;

    if (tid < 16) {
        int idx = tile * 16 + tid;
        if (idx < cnt) { s_tok[tid] = etok[e * CAP + idx]; s_wgt[tid] = ew[e * CAP + idx]; }
        else           { s_tok[tid] = 0;                   s_wgt[tid] = 0.f; }
    }
    __syncthreads();

    // ---- TDM descriptor issue: gather 16 token rows x 512 bf16 into LDS ----
    auto issue_panel = [&](int p, int buf) {
        unsigned lds_base = (unsigned)(uintptr_t)(const void*)&lx[buf][0];
        unsigned long long ga =
            (unsigned long long)(uintptr_t)xbf + (unsigned long long)p * (512 * 2);
        v4u g0;
        g0[0] = 1u | (0u << 30) | (1u << 31);       // count=1, 16-bit idx, gather_mode
        g0[1] = lds_base;                           // lds_addr (bytes)
        g0[2] = (unsigned)ga;                       // global_addr[31:0]
        g0[3] = (unsigned)((ga >> 32) & 0x01FFFFFFu) | (2u << 30); // addr[56:32] | type=2
        v8i_t g1;
        g1[0] = 1 << 16;                                   // data_size = 2 bytes
        g1[1] = (int)((512u & 0xffffu) << 16);             // tensor_dim0 lo16 (row width 512)
        g1[2] = (int)((512u >> 16) | ((unsigned)(T & 0xffff) << 16)); // td0 hi | td1 lo
        g1[3] = (int)(((unsigned)T >> 16) | (512u << 16)); // td1 hi | tile_dim0 = 512
        g1[4] = 16;                                        // tile_dim1 = 16 valid indices
        g1[5] = H;                                         // tensor_dim0_stride = 2048
        g1[6] = 0; g1[7] = 0;
        v4i_t g2, g3;
#pragma unroll
        for (int j = 0; j < 4; ++j) {
            g2[j] = s_tok[2 * j]     | (s_tok[2 * j + 1] << 16);
            g3[j] = s_tok[2 * j + 8] | (s_tok[2 * j + 9] << 16);
        }
        v8i_t gpad = {};                                   // unused 5th SGPR group (clang-23 form)
        __builtin_amdgcn_tensor_load_to_lds(g0, g1, g2, g3, gpad, 0);
    };

    if (wid == 0) issue_panel(0, 0);

    const unsigned short* w1e = w1bf + (size_t)e * F * H;
    const unsigned short* w3e = w3bf + (size_t)e * F * H;
    const unsigned short* w2e = w2bf + (size_t)e * H * F;

    const int nBase = wid * 32;                // this wave's F-column strip
    v8f g0a = {}, g1a = {}, u0a = {}, u1a = {};

    // ---- stage B: 4 K-panels of 512, DMA(p+1) overlapped with WMMA(p) ----
    for (int p = 0; p < 4; ++p) {
        __syncthreads();                       // panel p-1 consumed -> buf (p+1)&1 free
        if (wid == 0) {
            if (p < 3) {
                issue_panel(p + 1, (p + 1) & 1);
                __builtin_amdgcn_s_wait_tensorcnt(1);  // in-order: panel p done
            } else {
                __builtin_amdgcn_s_wait_tensorcnt(0);
            }
        }
        __syncthreads();                       // panel p visible to all waves
        const unsigned short* arow = &lx[p & 1][0] + l15 * 512;
        for (int kk = 0; kk < 16; ++kk) {
            const int kl = kk * 32;
            const int kg = p * 512 + kl;
            v16bf a = load_a_frag(arow, kl, chunk);
            const unsigned short* pG = w1e + (size_t)(nBase + l15) * H + kg + 16 * hw;
            const unsigned short* pU = w3e + (size_t)(nBase + l15) * H + kg + 16 * hw;
            g0a = wmma_bf16(a, load_b_frag(pG), g0a);
            g1a = wmma_bf16(a, load_b_frag(pG + (size_t)16 * H), g1a);
            u0a = wmma_bf16(a, load_b_frag(pU), u0a);
            u1a = wmma_bf16(a, load_b_frag(pU + (size_t)16 * H), u1a);
        }
    }

    // ---- stage C: h = silu(g) * u (fast v_rcp sigmoid), bf16 into LDS ----
    // C/D layout: VGPR i -> M = i + 8*hw, N = lane&15
#pragma unroll
    for (int i = 0; i < 8; ++i) {
        const int m = i + hw * 8;
        float g = g0a[i];
        float s = __builtin_amdgcn_rcpf(1.f + __expf(-g));
        lh[m * 256 + nBase + l15] = f2bf(g * s * u0a[i]);
        g = g1a[i];
        s = __builtin_amdgcn_rcpf(1.f + __expf(-g));
        lh[m * 256 + nBase + 16 + l15] = f2bf(g * s * u1a[i]);
    }
    __syncthreads();

    // ---- stage D: out_tile = h @ w2^T, scale, scatter-add ----
    const unsigned short* hrow = lh + l15 * 256;
    const int colW = wid * 256;                // this wave's H-column strip
    for (int t = 0; t < 16; ++t) {
        const int col0 = colW + t * 16;
        v8f acc = {};
#pragma unroll
        for (int kk = 0; kk < 8; ++kk) {
            v16bf a = load_a_frag(hrow, kk * 32, chunk);
            const unsigned short* pB = w2e + (size_t)(col0 + l15) * F + kk * 32 + 16 * hw;
            acc = wmma_bf16(a, load_b_frag(pB), acc);
        }
        const int col = col0 + l15;
#pragma unroll
        for (int i = 0; i < 8; ++i) {
            const int m = i + hw * 8;
            const float wv = s_wgt[m];
            if (wv > 0.f)
                atomicAdd(out + (size_t)s_tok[m] * H + col, acc[i] * wv);
        }
    }
}

// ---------------- host launch ----------------
extern "C" void kernel_launch(void* const* d_in, const int* in_sizes, int n_in,
                              void* d_out, int out_size, void* d_ws, size_t ws_size,
                              hipStream_t stream) {
    (void)in_sizes; (void)n_in; (void)out_size; (void)ws_size;
    const float* x  = (const float*)d_in[0];
    const float* gw = (const float*)d_in[1];
    const float* w1 = (const float*)d_in[2];
    const float* w2 = (const float*)d_in[3];
    const float* w3 = (const float*)d_in[4];
    float* out = (float*)d_out;

    // workspace layout (~102 MB): bf16 copies first (32B aligned), then routing
    unsigned short* xbf  = (unsigned short*)d_ws;          // T*H
    unsigned short* w1bf = xbf  + (size_t)T * H;           // E*F*H
    unsigned short* w2bf = w1bf + (size_t)E * F * H;
    unsigned short* w3bf = w2bf + (size_t)E * F * H;
    int*   sel  = (int*)(w3bf + (size_t)E * F * H);        // T*2
    float* rw   = (float*)(sel + T * 2);                   // T*2
    int*   etok = (int*)(rw + T * 2);                      // E*CAP
    float* ew   = (float*)(etok + E * CAP);                // E*CAP
    int*   ecnt = (int*)(ew + E * CAP);                    // E

    (void)hipMemsetAsync(d_out, 0, (size_t)T * H * sizeof(float), stream);

    cvt_bf16_kernel<<<((size_t)T * H) / 2048, 256, 0, stream>>>(x,  xbf);
    cvt_bf16_kernel<<<((size_t)E * F * H) / 2048, 256, 0, stream>>>(w1, w1bf);
    cvt_bf16_kernel<<<((size_t)E * F * H) / 2048, 256, 0, stream>>>(w2, w2bf);
    cvt_bf16_kernel<<<((size_t)E * F * H) / 2048, 256, 0, stream>>>(w3, w3bf);

    gate_kernel <<<T / 8, 256, 0, stream>>>(x, gw, sel, rw);
    route_kernel<<<E, 1024, 0, stream>>>(sel, rw, etok, ew, ecnt);
    ffn_kernel  <<<E * TILES, 256, 0, stream>>>(xbf, w1bf, w2bf, w3bf,
                                                etok, ew, ecnt, out);
}
